// MultiSimilarityLoss_78185584656815
// MI455X (gfx1250) — matrix operand order repaired and checked by reading, not compiled, not run on hardware
//
#include <hip/hip_runtime.h>
#include <hip/hip_bf16.h>

// Multi-similarity loss on MI455X (gfx1250).
// sim = E @ E^T via split-precision bf16x3 WMMA (v_wmma_f32_16x16x32_bf16):
//   E = Ehi + Elo (bf16 each), sim ~= Ehi.Ehi^T + Ehi.Elo^T + Elo.Ehi^T
// (error ~2^-16, safe vs. the 0.1 mining margins and exp(50*sim) scaling).
// 16-bit WMMA issues as TRANS-class, co-executing with the mining VALU work.
// Two WMMA sweeps per 16-row block:
//   sweep 1: collect positive-pair sims into LDS (sparse) + min_pos
//   sweep 2: max mined negative + neg exp-sum
// then finalize positives from the LDS list (no 3rd GEMM).

typedef __bf16 v8bf  __attribute__((ext_vector_type(8)));
typedef __bf16 v16bf __attribute__((ext_vector_type(16)));
typedef float  v8f   __attribute__((ext_vector_type(8)));

#define B_N     8192
#define D_K     128
#define KCHUNK  4              // 4 chunks of K=32 per 16x16 tile
#define NTILES  (B_N / 16)     // 512 column tiles
#define NBLK    (B_N / 16)     // 512 row blocks
#define NWAVES  4
#define POSCAP  256

#define ALPHA_F   2.0f
#define BETA_F    50.0f
#define BASE_F    0.5f
#define INV_ALPHA 0.5f
#define INV_BETA  0.02f

__device__ __forceinline__ v8f wmma_bf16(v16bf a, v16bf b, v8f c) {
  // (neg_a, A, neg_b, B, c_mod, C, reuse_a, reuse_b)
  return __builtin_amdgcn_wmma_f32_16x16x32_bf16(false, a, false, b, (short)0,
                                                 c, false, false);
}

__device__ __forceinline__ unsigned short f32_to_bf16_rne(float x) {
  unsigned u = __float_as_uint(x);
  return (unsigned short)((u + 0x7FFFu + ((u >> 16) & 1u)) >> 16);
}

// ---- Split each f32 embedding element into hi/lo bf16 (bf16x3 scheme) ----
__global__ __launch_bounds__(256) void
ms_split_bf16(const float* __restrict__ emb, unsigned short* __restrict__ hi,
              unsigned short* __restrict__ lo) {
  const int idx = blockIdx.x * 256 + threadIdx.x;   // B_N*D_K total
  const float x = emb[idx];
  const unsigned short hb = f32_to_bf16_rne(x);
  hi[idx] = hb;
  const float hf = __uint_as_float(((unsigned)hb) << 16);
  lo[idx] = f32_to_bf16_rne(x - hf);
}

__global__ __launch_bounds__(128) void
ms_loss_main(const unsigned short* __restrict__ hi_u,
             const unsigned short* __restrict__ lo_u,
             const int* __restrict__ labels, float* __restrict__ accum) {
  __shared__ float s_pos[16][POSCAP];   // collected positive-pair sims per row
  __shared__ int   s_cnt[16];
  __shared__ float s_redA[NWAVES][16];
  __shared__ float s_redB[NWAVES][16];
  __shared__ float s_minpos[16];
  __shared__ float s_maxneg[16];
  __shared__ float s_negsum[16];
  __shared__ float s_loss[16];
  __shared__ float s_vcnt[16];

  const __bf16* __restrict__ hi = (const __bf16*)hi_u;
  const __bf16* __restrict__ lo = (const __bf16*)lo_u;

  const int tid  = threadIdx.x;
  const int wave = tid >> 5;
  const int lane = tid & 31;
  const int half = lane >> 4;          // 0: lanes 0-15, 1: lanes 16-31
  const int l16  = lane & 15;

  const int row_base = blockIdx.x * 16;

  if (tid < 16) s_cnt[tid] = 0;
  __syncthreads();

  // ---- Preload A operands (hi & lo): rows [row_base, row_base+16) ----
  // A 16x32 order used here (not ISA table order): lane holds M=l%16; comps 0-15 = K = 16*half+{0..15}
  // (same contiguous order as B).
  const int ia = row_base + l16;
  const __bf16* ahrow = hi + (size_t)ia * D_K + 16 * half;
  const __bf16* alrow = lo + (size_t)ia * D_K + 16 * half;
  v16bf a_hi[KCHUNK], a_lo[KCHUNK];
#pragma unroll
  for (int c = 0; c < KCHUNK; ++c) {
    v8bf h0 = *(const v8bf*)(ahrow + 32 * c);
    v8bf h1 = *(const v8bf*)(ahrow + 32 * c + 8);
    a_hi[c] = __builtin_shufflevector(h0, h1, 0, 1, 2, 3, 4, 5, 6, 7,
                                      8, 9, 10, 11, 12, 13, 14, 15);
    v8bf l0 = *(const v8bf*)(alrow + 32 * c);
    v8bf l1 = *(const v8bf*)(alrow + 32 * c + 8);
    a_lo[c] = __builtin_shufflevector(l0, l1, 0, 1, 2, 3, 4, 5, 6, 7,
                                      8, 9, 10, 11, 12, 13, 14, 15);
  }

  // Row labels / indices for the C-tile rows this lane touches:
  // C VGPR r, lane l -> (M = r + 8*(l>>4), N = l&15)
  int labrow[8], irow[8];
#pragma unroll
  for (int r = 0; r < 8; ++r) {
    irow[r]   = row_base + r + 8 * half;
    labrow[r] = labels[irow[r]];
  }

  // ================= Sweep 1: collect positives + min_pos =================
  float lmin[8];
#pragma unroll
  for (int r = 0; r < 8; ++r) lmin[r] = INFINITY;

  for (int t = wave; t < NTILES; t += NWAVES) {
    const int jn = t * 16 + l16;
    const int lab_j = labels[jn];
    // B 32x16 order used here (matches A): lane holds N=l%16; comps = K = 16*half+{0..15}.
    const __bf16* bh = hi + (size_t)jn * D_K + 16 * half;
    const __bf16* bl = lo + (size_t)jn * D_K + 16 * half;

    v8f chh = {}, chl = {}, clh = {};
#pragma unroll
    for (int c = 0; c < KCHUNK; ++c) {
      v16bf bhi = *(const v16bf*)(bh + 32 * c);
      v16bf blo = *(const v16bf*)(bl + 32 * c);
      chh = wmma_bf16(a_hi[c], bhi, chh);
      chl = wmma_bf16(a_hi[c], blo, chl);
      clh = wmma_bf16(a_lo[c], bhi, clh);
    }
#pragma unroll
    for (int r = 0; r < 8; ++r) {
      float sim = chh[r] + (chl[r] + clh[r]);
      if ((labrow[r] == lab_j) && (irow[r] != jn)) {   // positive pair
        lmin[r] = fminf(lmin[r], sim);
        int M = r + 8 * half;
        int idx = atomicAdd(&s_cnt[M], 1);
        if (idx < POSCAP) s_pos[M][idx] = sim;
      }
    }
  }

  // Reduce min over the 16 lanes sharing each row, combine across waves.
#pragma unroll
  for (int r = 0; r < 8; ++r) {
    float v = lmin[r];
#pragma unroll
    for (int m = 8; m >= 1; m >>= 1) v = fminf(v, __shfl_xor(v, m, 32));
    if (l16 == 0) s_redA[wave][r + 8 * half] = v;
  }
  __syncthreads();
  if (tid < 16) {
    float v = fminf(fminf(s_redA[0][tid], s_redA[1][tid]),
                    fminf(s_redA[2][tid], s_redA[3][tid]));
    s_minpos[tid] = v;
  }
  __syncthreads();

  // ================= Sweep 2: mined negatives (max + exp-sum) =============
  float thn[8];
#pragma unroll
  for (int r = 0; r < 8; ++r) thn[r] = s_minpos[r + 8 * half] - 0.1f;

  float lmax[8], lsum[8];
#pragma unroll
  for (int r = 0; r < 8; ++r) { lmax[r] = -INFINITY; lsum[r] = 0.f; }

  for (int t = wave; t < NTILES; t += NWAVES) {
    const int jn = t * 16 + l16;
    const int lab_j = labels[jn];
    const __bf16* bh = hi + (size_t)jn * D_K + 16 * half;
    const __bf16* bl = lo + (size_t)jn * D_K + 16 * half;

    v8f chh = {}, chl = {}, clh = {};
#pragma unroll
    for (int c = 0; c < KCHUNK; ++c) {
      v16bf bhi = *(const v16bf*)(bh + 32 * c);
      v16bf blo = *(const v16bf*)(bl + 32 * c);
      chh = wmma_bf16(a_hi[c], bhi, chh);
      chl = wmma_bf16(a_hi[c], blo, chl);
      clh = wmma_bf16(a_lo[c], bhi, clh);
    }
#pragma unroll
    for (int r = 0; r < 8; ++r) {
      float sim = chh[r] + (chl[r] + clh[r]);
      if ((labrow[r] != lab_j) && (sim > thn[r])) {    // mined negative
        lmax[r] = fmaxf(lmax[r], sim);
        lsum[r] += __expf(BETA_F * (sim - BASE_F));
      }
    }
  }

#pragma unroll
  for (int r = 0; r < 8; ++r) {
    float vm = lmax[r], vs = lsum[r];
#pragma unroll
    for (int m = 8; m >= 1; m >>= 1) {
      vm = fmaxf(vm, __shfl_xor(vm, m, 32));
      vs += __shfl_xor(vs, m, 32);
    }
    if (l16 == 0) {
      s_redA[wave][r + 8 * half] = vm;
      s_redB[wave][r + 8 * half] = vs;
    }
  }
  __syncthreads();
  if (tid < 16) {
    s_maxneg[tid] = fmaxf(fmaxf(s_redA[0][tid], s_redA[1][tid]),
                          fmaxf(s_redA[2][tid], s_redA[3][tid]));
    s_negsum[tid] = s_redB[0][tid] + s_redB[1][tid] +
                    s_redB[2][tid] + s_redB[3][tid];
  }
  __syncthreads();

  // ================= Finalize rows from collected positives ===============
  if (tid < 16) {
    const int   M   = tid;
    const float mn  = s_maxneg[M];
    const float ns  = s_negsum[M];
    const float thp = mn + 0.1f;
    const int   cnt = min(s_cnt[M], POSCAP);
    float ps = 0.f;
    bool anyp = false;
    for (int k = 0; k < cnt; ++k) {
      float s = s_pos[M][k];
      if (s < thp) { anyp = true; ps += __expf(-ALPHA_F * (s - BASE_F)); }
    }
    const bool anyn  = (mn > -INFINITY);
    const bool valid = anyp && anyn;
    s_loss[M] = valid ? (log1pf(ps) * INV_ALPHA + log1pf(ns) * INV_BETA) : 0.f;
    s_vcnt[M] = valid ? 1.f : 0.f;
  }
  __syncthreads();
  if (tid == 0) {
    float ls = 0.f, vc = 0.f;
#pragma unroll
    for (int r = 0; r < 16; ++r) { ls += s_loss[r]; vc += s_vcnt[r]; }
    atomicAdd(&accum[0], ls);
    atomicAdd(&accum[1], vc);
  }
}

__global__ void ms_loss_init(float* __restrict__ accum) {
  if (threadIdx.x < 2) accum[threadIdx.x] = 0.f;
}

__global__ void ms_loss_final(const float* __restrict__ accum,
                              float* __restrict__ out) {
  if (threadIdx.x == 0) out[0] = accum[0] / fmaxf(accum[1], 1.f);
}

extern "C" void kernel_launch(void* const* d_in, const int* in_sizes, int n_in,
                              void* d_out, int out_size, void* d_ws, size_t ws_size,
                              hipStream_t stream) {
  const float* emb    = (const float*)d_in[0];
  const int*   labels = (const int*)d_in[1];
  float*       out    = (float*)d_out;

  // Workspace layout: [0,1]: accumulators (loss sum, valid count);
  // offset 1KB: hi bf16 array (2 MB); then lo bf16 array (2 MB).
  float*          accum = (float*)d_ws;
  unsigned short* hi    = (unsigned short*)((char*)d_ws + 1024);
  unsigned short* lo    = hi + (size_t)B_N * D_K;

  ms_split_bf16<<<(B_N * D_K) / 256, 256, 0, stream>>>(emb, hi, lo);
  ms_loss_init<<<1, 32, 0, stream>>>(accum);
  ms_loss_main<<<NBLK, NWAVES * 32, 0, stream>>>(hi, lo, labels, accum);
  ms_loss_final<<<1, 32, 0, stream>>>(accum, out);
}
